// Triangle_Update_Outgoing_34660386079030
// MI455X (gfx1250) — matrix operand order repaired and checked by reading, not compiled, not run on hardware
//
#include <hip/hip_runtime.h>
#include <hip/hip_bf16.h>
#include <stdint.h>

// ---------------------------------------------------------------------------
// Triangle multiplicative update (outgoing), MI455X / gfx1250, wave32 + WMMA
// B=1, N=512, C=H=128
// ---------------------------------------------------------------------------

#define NN     512
#define CC     128
#define HH     128
#define RR     (NN * NN)          // 262144 flattened (i,j) rows
#define EPSV   1e-5f

typedef __attribute__((ext_vector_type(16))) __bf16 v16bf;
typedef __attribute__((ext_vector_type(8)))  float  v8f;
typedef __attribute__((ext_vector_type(4)))  int    vi4;

union FragB { v16bf v; uint4 q[2]; };            // 16 bf16 = 32 B = 8 VGPRs
union AccU  { v8f   v; float f[8]; };            // 16x16 f32 C/D tile slice
union Pack8 { unsigned short s[8]; uint4 q; };   // 8 bf16 packed for b128 store

__device__ __forceinline__ unsigned short f2bf(float f) {
    unsigned int u = __float_as_uint(f);
    u += 0x7FFFu + ((u >> 16) & 1u);             // round-to-nearest-even
    return (unsigned short)(u >> 16);
}
__device__ __forceinline__ float bf2f(unsigned short h) {
    return __uint_as_float(((unsigned int)h) << 16);
}
__device__ __forceinline__ float sigmoidf_(float x) {
    return 1.0f / (1.0f + __expf(-x));
}

// --------------------- CDNA5 async global->LDS path ------------------------
#define ASG __attribute__((address_space(1)))
#define ASL __attribute__((address_space(3)))

#if __has_builtin(__builtin_amdgcn_global_load_async_to_lds_b128)
#define HAVE_ASYNC_LDS 1
__device__ __forceinline__ void async_cp16(const void* g, void* l) {
    __builtin_amdgcn_global_load_async_to_lds_b128(
        (ASG vi4*)(g), (ASL vi4*)(l), 0, 0);
}
__device__ __forceinline__ void wait_async0() {
#if __has_builtin(__builtin_amdgcn_s_wait_asynccnt)
    __builtin_amdgcn_s_wait_asynccnt(0);
#else
    asm volatile("s_wait_asynccnt 0x0" ::: "memory");
#endif
}
#endif

// ---------------------------------------------------------------------------
// Kernel P: convert the six 128x128 weight matrices to bf16 (kept [out][in]
// so WMMA B-fragments read two 16 B k-runs per lane).
// Order in ws: Wa, Ga, Wb, Gb, Wg, Wo
// ---------------------------------------------------------------------------
__global__ __launch_bounds__(256) void prep_weights(
    const float* __restrict__ Wa, const float* __restrict__ Ga,
    const float* __restrict__ Wb, const float* __restrict__ Gb,
    const float* __restrict__ Wg, const float* __restrict__ Wo,
    unsigned short* __restrict__ wbf)
{
    int idx = blockIdx.x * blockDim.x + threadIdx.x;   // 0 .. 6*16384
    if (idx >= 6 * 16384) return;
    const float* srcs[6] = { Wa, Ga, Wb, Gb, Wg, Wo };
    int m = idx >> 14, e = idx & 16383;
    wbf[idx] = f2bf(srcs[m][e]);
}

// ---------------------------------------------------------------------------
// Kernel A: fused LayerNorm(C) + 5 projections + gating.
// Block = 256 threads (8 wave32) handles 16 consecutive flattened rows.
// Wave w owns output columns h in [16w, 16w+16) and computes Wa,Ga,Wb,Gb,Wg
// sharing the same z A-fragments. a,b stored TRANSPOSED bf16: at[h][i][k=j],
// bt[h][j][k] -> per-lane 16B contiguous stores from the WMMA D layout.
// ---------------------------------------------------------------------------
__global__ __launch_bounds__(256) void ln_proj_kernel(
    const float* __restrict__ pair,
    const float* __restrict__ ln_g, const float* __restrict__ ln_b,
    const unsigned short* __restrict__ wbf,
    const float* __restrict__ ba,  const float* __restrict__ bga,
    const float* __restrict__ bb,  const float* __restrict__ bgb,
    const float* __restrict__ bg,
    unsigned short* __restrict__ at, unsigned short* __restrict__ bt,
    unsigned short* __restrict__ gbf)
{
    __shared__ unsigned short zsh[16 * 136];   // 16 rows x 128 bf16, pad->136

    const int t   = threadIdx.x;
    const int r0  = blockIdx.x * 16;           // 16 flattened rows, same i
    const int row = t >> 4;                    // 0..15
    const int seg = t & 15;                    // 8-col segment

    // ---- LayerNorm over C (16 lanes per row, shfl_xor reduction) ----
    const float* prow = pair + (size_t)(r0 + row) * CC + seg * 8;
    float4 a0 = *(const float4*)(prow);
    float4 a1 = *(const float4*)(prow + 4);
    float x[8] = { a0.x, a0.y, a0.z, a0.w, a1.x, a1.y, a1.z, a1.w };

    float s = 0.f, s2 = 0.f;
#pragma unroll
    for (int u = 0; u < 8; ++u) { s += x[u]; s2 += x[u] * x[u]; }
#pragma unroll
    for (int m = 8; m >= 1; m >>= 1) {
        s  += __shfl_xor(s,  m, 16);
        s2 += __shfl_xor(s2, m, 16);
    }
    const float mu   = s  * (1.0f / CC);
    const float var  = s2 * (1.0f / CC) - mu * mu;
    const float rstd = rsqrtf(var + EPSV);

    Pack8 zp;
#pragma unroll
    for (int u = 0; u < 8; ++u) {
        int c = seg * 8 + u;
        zp.s[u] = f2bf((x[u] - mu) * rstd * ln_g[c] + ln_b[c]);
    }
    *(uint4*)&zsh[row * 136 + seg * 8] = zp.q;
    __syncthreads();

    // ---- WMMA projections ----
    const int w    = t >> 5;        // wave 0..7 -> h-tile
    const int l    = t & 31;
    const int ln15 = l & 15;
    const int lhi  = l >> 4;
    const int h0   = w * 16;

    FragB A[4];
#pragma unroll
    for (int kk = 0; kk < 4; ++kk) {
        const unsigned short* p = &zsh[ln15 * 136 + kk * 32 + lhi * 8];
        A[kk].q[0] = *(const uint4*)(p);
        A[kk].q[1] = *(const uint4*)(p + 16);   // K + 16
    }

    AccU P[5];
#pragma unroll
    for (int m = 0; m < 5; ++m) {
        v8f acc = {};
        const unsigned short* wm = wbf + m * 16384 + (h0 + ln15) * CC;
#pragma unroll
        for (int kk = 0; kk < 4; ++kk) {
            FragB B;
            const unsigned short* p = wm + kk * 32 + lhi * 8;
            B.q[0] = *(const uint4*)(p);
            B.q[1] = *(const uint4*)(p + 16);
            acc = __builtin_amdgcn_wmma_f32_16x16x32_bf16(
                false, A[kk].v, false, B.v, (short)0, acc, false, false);
        }
        P[m].v = acc;
    }

    // ---- gating + transposed stores ----
    const int i  = r0 >> 9;
    const int j0 = r0 & 511;
    const float bav  = ba [h0 + ln15];
    const float bgav = bga[h0 + ln15];
    const float bbv  = bb [h0 + ln15];
    const float bgbv = bgb[h0 + ln15];
    const float bgv  = bg [h0 + ln15];

    Pack8 apk, bpk;
#pragma unroll
    for (int v = 0; v < 8; ++v) {
        float av = (P[0].f[v] + bav) * sigmoidf_(P[1].f[v] + bgav);
        float bv = (P[2].f[v] + bbv) * sigmoidf_(P[3].f[v] + bgbv);
        apk.s[v] = f2bf(av);
        bpk.s[v] = f2bf(bv);
    }
    const size_t tbase = (size_t)(h0 + ln15) * RR + (size_t)i * NN + j0 + lhi * 8;
    *(uint4*)&at[tbase] = apk.q;
    *(uint4*)&bt[tbase] = bpk.q;

#pragma unroll
    for (int v = 0; v < 8; ++v) {
        int m = v + lhi * 8;
        gbf[(size_t)(r0 + m) * CC + h0 + ln15] = f2bf(sigmoidf_(P[4].f[v] + bgv));
    }
}

// ---------------------------------------------------------------------------
// Kernel B: triangle einsum t[i,j,h] = sum_k a[i,k,h] b[j,k,h].
// One block = one h-plane x one 128x128 (i,j) macro-tile -> re-read factor of
// at/bt drops to 4x (~0.5 GB of L2 traffic instead of 4.3 GB).
// Double-buffered LDS panels (row pitch 40 halfs = 80 B: b128 fragment reads
// are 16B-aligned and bank-conflict-free). k-chunk staging is software-
// pipelined; uses CDNA5 async global->LDS (ASYNCcnt) when available.
// Wave w (wi=w>>1, wj=w&1) owns a 32x64 patch = 2x4 WMMA tiles.
// ---------------------------------------------------------------------------
#define APITCH 40                              // halfs per LDS panel row

__device__ __forceinline__ void stage_chunk(
    const unsigned short* __restrict__ ap, const unsigned short* __restrict__ bp,
    unsigned short* aL, unsigned short* bL, int t, int k0)
{
    const int r = t >> 1, half = t & 1;        // 128 rows, 2 x 32B per row
    const unsigned short* ga = ap + (size_t)r * NN + k0 + half * 16;
    const unsigned short* gb = bp + (size_t)r * NN + k0 + half * 16;
    unsigned short* la = aL + r * APITCH + half * 16;
    unsigned short* lb = bL + r * APITCH + half * 16;
#ifdef HAVE_ASYNC_LDS
    async_cp16(ga,     la);
    async_cp16(ga + 8, la + 8);
    async_cp16(gb,     lb);
    async_cp16(gb + 8, lb + 8);
#else
    uint4 a0 = *(const uint4*)(ga), a1 = *(const uint4*)(ga + 8);
    uint4 b0 = *(const uint4*)(gb), b1 = *(const uint4*)(gb + 8);
    *(uint4*)(la) = a0; *(uint4*)(la + 8) = a1;
    *(uint4*)(lb) = b0; *(uint4*)(lb + 8) = b1;
#endif
}

__device__ __forceinline__ void stage_commit() {
#ifdef HAVE_ASYNC_LDS
    wait_async0();
#endif
    __syncthreads();
}

__global__ __launch_bounds__(256) void tri_einsum_kernel(
    const unsigned short* __restrict__ at, const unsigned short* __restrict__ bt,
    float* __restrict__ tfp)
{
    __shared__ unsigned short aL[2][128 * APITCH];   // 2 x 10 KB
    __shared__ unsigned short bL[2][128 * APITCH];   // 2 x 10 KB

    const int t    = threadIdx.x;
    const int w    = t >> 5;
    const int l    = t & 31;
    const int ln15 = l & 15;
    const int lhi  = l >> 4;
    const int wi   = w >> 1;                   // 0..3 -> i strip of 32
    const int wj   = w & 1;                    // 0..1 -> j strip of 64

    const int h  = blockIdx.y;
    const int i0 = (blockIdx.x >> 2) * 128;
    const int j0 = (blockIdx.x & 3) * 128;
    const size_t hb = (size_t)h * RR;
    const unsigned short* ap = at + hb + (size_t)i0 * NN;
    const unsigned short* bp = bt + hb + (size_t)j0 * NN;

    v8f acc[2][4] = {};

    stage_chunk(ap, bp, aL[0], bL[0], t, 0);
    stage_commit();

    for (int kk = 0; kk < 16; ++kk) {
        const int cur = kk & 1;
        if (kk + 1 < 16)
            stage_chunk(ap, bp, aL[cur ^ 1], bL[cur ^ 1], t, (kk + 1) * 32);

        FragB Bf[4];
#pragma unroll
        for (int nj = 0; nj < 4; ++nj) {
            const unsigned short* p =
                &bL[cur][(wj * 64 + nj * 16 + ln15) * APITCH + lhi * 8];
            Bf[nj].q[0] = *(const uint4*)(p);
            Bf[nj].q[1] = *(const uint4*)(p + 16);
        }
#pragma unroll
        for (int mi = 0; mi < 2; ++mi) {
            FragB Af;
            const unsigned short* p =
                &aL[cur][(wi * 32 + mi * 16 + ln15) * APITCH + lhi * 8];
            Af.q[0] = *(const uint4*)(p);
            Af.q[1] = *(const uint4*)(p + 16);
#pragma unroll
            for (int nj = 0; nj < 4; ++nj) {
                acc[mi][nj] = __builtin_amdgcn_wmma_f32_16x16x32_bf16(
                    false, Af.v, false, Bf[nj].v, (short)0, acc[mi][nj],
                    false, false);
            }
        }
        stage_commit();
    }

    // store t[h][i][j] f32 (64B contiguous per VGPR row across lanes 0-15)
#pragma unroll
    for (int mi = 0; mi < 2; ++mi) {
#pragma unroll
        for (int nj = 0; nj < 4; ++nj) {
            AccU u; u.v = acc[mi][nj];
            const int jb = j0 + wj * 64 + nj * 16 + ln15;
#pragma unroll
            for (int v = 0; v < 8; ++v) {
                const int gi = i0 + wi * 32 + mi * 16 + v + lhi * 8;
                tfp[hb + (size_t)gi * NN + jb] = u.f[v];
            }
        }
    }
}

// ---------------------------------------------------------------------------
// Kernel C: fused LayerNorm(H) + output projection + gate.
// Block = 256 threads owns one 16x16 (i,j) tile.
//  Stage 1: gather the 128 h-planes of t into LDS cube tacc[256][129]
//           (pad 129 -> conflict-free per-thread LN over contiguous h).
//  Stage 2: per-thread LN over h, bf16 restage tln[256][136].
//  Stage 3: WMMA projection with Wo (wave w owns c-tile w), +bo, *g_o, store.
// Dynamic LDS: 256*129*4 + 256*136*2 = 201728 B  (<320KB/WGP).
// ---------------------------------------------------------------------------
__global__ __launch_bounds__(256) void tri_ln_out_kernel(
    const float* __restrict__ tfp,
    const unsigned short* __restrict__ gbf, const unsigned short* __restrict__ wobf,
    const float* __restrict__ lnh_g, const float* __restrict__ lnh_b,
    const float* __restrict__ bo,
    float* __restrict__ out)
{
    extern __shared__ char smem[];
    float*          tacc = (float*)smem;                          // [256][129] f32
    unsigned short* tln  = (unsigned short*)(smem + 256*129*4);   // [256][136] bf16
    __shared__ float lnhg[HH], lnhb[HH];

    const int t    = threadIdx.x;
    const int w    = t >> 5;
    const int l    = t & 31;
    const int ln15 = l & 15;
    const int lhi  = l >> 4;
    const int i0   = blockIdx.y * 16;
    const int j0   = blockIdx.x * 16;

    if (t < HH) { lnhg[t] = lnh_g[t]; lnhb[t] = lnh_b[t]; }

    // ---- Stage 1: gather t over h (thread t <-> element mn = t) ----
    {
        const float* src = tfp + (size_t)(i0 + (t >> 4)) * NN + (j0 + (t & 15));
        float* dst = &tacc[t * 129];
#pragma unroll 4
        for (int h = 0; h < HH; ++h)
            dst[h] = src[(size_t)h * RR];
    }
    __syncthreads();

    // ---- Stage 2: LayerNorm over h ----
    {
        const float* p = &tacc[t * 129];
        float s = 0.f, s2 = 0.f;
#pragma unroll 8
        for (int h = 0; h < HH; ++h) { float v = p[h]; s += v; s2 += v * v; }
        const float mu   = s  * (1.0f / HH);
        const float var  = s2 * (1.0f / HH) - mu * mu;
        const float rstd = rsqrtf(var + EPSV);
        unsigned short* q = &tln[t * 136];
#pragma unroll 8
        for (int h = 0; h < HH; ++h)
            q[h] = f2bf((p[h] - mu) * rstd * lnhg[h] + lnhb[h]);
    }
    __syncthreads();

    // ---- Stage 3: out = (t_ln @ Wo^T + bo) * g_o ----
    const int   c0  = w * 16;
    const float bov = bo[c0 + ln15];
#pragma unroll 1
    for (int mt = 0; mt < 16; ++mt) {             // M-tiles: i_local = mt
        v8f acc = {};
#pragma unroll
        for (int kk = 0; kk < 4; ++kk) {
            FragB A, B;
            const unsigned short* pa = &tln[(mt * 16 + ln15) * 136 + kk * 32 + lhi * 8];
            A.q[0] = *(const uint4*)(pa);
            A.q[1] = *(const uint4*)(pa + 16);
            const unsigned short* pb = wobf + (c0 + ln15) * HH + kk * 32 + lhi * 8;
            B.q[0] = *(const uint4*)(pb);
            B.q[1] = *(const uint4*)(pb + 16);
            acc = __builtin_amdgcn_wmma_f32_16x16x32_bf16(
                false, A.v, false, B.v, (short)0, acc, false, false);
        }
        AccU u; u.v = acc;
        const int gi = i0 + mt;
#pragma unroll
        for (int v = 0; v < 8; ++v) {
            int m = v + lhi * 8;                  // j_local
            size_t r = (size_t)gi * NN + j0 + m;
            float g = bf2f(gbf[r * CC + c0 + ln15]);
            out[r * CC + c0 + ln15] = (u.f[v] + bov) * g;
        }
    }
}

// ---------------------------------------------------------------------------
// Workspace layout (bytes):
//   [0)          wbf : 6*16384*2        = 196608
//   [196608)     at  : 128*512*512*2    = 67108864   (a transposed, bf16)
//   [+at]        bt  : 67108864
//   [+bt]        gbf : 67108864                      (g_o, bf16)
//   [+gbf]       t   : 128*512*512*4    = 134217728  (einsum result, f32)
//   total ~320.2 MiB
// ---------------------------------------------------------------------------
extern "C" void kernel_launch(void* const* d_in, const int* in_sizes, int n_in,
                              void* d_out, int out_size, void* d_ws, size_t ws_size,
                              hipStream_t stream)
{
    const float* pair  = (const float*)d_in[0];
    const float* ln_g  = (const float*)d_in[1];
    const float* ln_b  = (const float*)d_in[2];
    const float* lnh_g = (const float*)d_in[3];
    const float* lnh_b = (const float*)d_in[4];
    const float* Wa    = (const float*)d_in[5];
    const float* ba    = (const float*)d_in[6];
    const float* Ga    = (const float*)d_in[7];
    const float* bga   = (const float*)d_in[8];
    const float* Wb    = (const float*)d_in[9];
    const float* bb    = (const float*)d_in[10];
    const float* Gb    = (const float*)d_in[11];
    const float* bgb   = (const float*)d_in[12];
    const float* Wo    = (const float*)d_in[13];
    const float* bo    = (const float*)d_in[14];
    const float* Wg    = (const float*)d_in[15];
    const float* bg    = (const float*)d_in[16];

    char* ws = (char*)d_ws;
    unsigned short* wbf  = (unsigned short*)ws;
    unsigned short* at   = (unsigned short*)(ws + 196608);
    unsigned short* bt   = (unsigned short*)(ws + 196608 + (size_t)67108864);
    unsigned short* gbf  = (unsigned short*)(ws + 196608 + 2 * (size_t)67108864);
    float*          tfp  = (float*)        (ws + 196608 + 3 * (size_t)67108864);
    unsigned short* wobf = wbf + 5 * 16384;

    const size_t lds_c = 256 * 129 * 4 + 256 * 136 * 2;   // 201728 B
    (void)hipFuncSetAttribute((const void*)tri_ln_out_kernel,
                              hipFuncAttributeMaxDynamicSharedMemorySize,
                              (int)lds_c);

    prep_weights<<<(6 * 16384 + 255) / 256, 256, 0, stream>>>(
        Wa, Ga, Wb, Gb, Wg, Wo, wbf);

    ln_proj_kernel<<<RR / 16, 256, 0, stream>>>(
        pair, ln_g, ln_b, wbf, ba, bga, bb, bgb, bg, at, bt, gbf);

    dim3 grid_b(16, 128);    // 16 macro-tiles x 128 h-planes
    tri_einsum_kernel<<<grid_b, 256, 0, stream>>>(at, bt, tfp);

    dim3 grid_c(NN / 16, NN / 16);
    tri_ln_out_kernel<<<grid_c, 256, lds_c, stream>>>(
        tfp, gbf, wobf, lnh_g, lnh_b, bo, (float*)d_out);
}
